// BilinearAttnTransform_7748121002184
// MI455X (gfx1250) — matrix operand order repaired and checked by reading, not compiled, not run on hardware
//
#include <hip/hip_runtime.h>
#include <hip/hip_bf16.h>
#include <math.h>

// ---------------------------------------------------------------------------
// BilinearAttnTransform on MI455X (gfx1250, wave32, WMMA)
//   B=16, C=128, H=W=128, G=4, BS=8, cg=32, th=tw=16, OG=256
// Stage 1: conv1(1x1,C->G)+BN+ReLU fused with block max-pools (never
//          materializes the G-channel map).  fp32 VALU, bandwidth bound.
// Stage 2: tiny p/q sigmoid attention + row/col normalization.
// Stage 3: bilinear block mixing through LDS, emits z as bf16 with packed
//          2xbf16 u32 stores (128B contiguous per wave).
// Stage 4: conv2 as per-batch (128x128)x(128x16384) GEMM with
//          v_wmma_f32_16x16x32_bf16 (fp32 accum), BN+ReLU epilogue.
//          z (67MB) stays resident in the 192MB L2 between stages 3 and 4.
// ---------------------------------------------------------------------------

#define Bn   16
#define Cn   128
#define Gn   4
#define CG   32
#define HW   16384   // 128*128
#define Wd   128
#define EPSV 1e-5f

typedef __attribute__((ext_vector_type(16))) __bf16 v16bf;
typedef __attribute__((ext_vector_type(8)))  float  v8f;

union FragAB { v16bf v; float4 f[2]; };

__device__ __forceinline__ unsigned int f32_to_bf16_bits(float x) {
    unsigned int u = __float_as_uint(x);
    return (u + 0x7FFFu + ((u >> 16) & 1u)) >> 16;
}

// ---------------------------------------------------------------------------
// Kernel 1: conv1 + BN + ReLU fused with block-max pooling.
// One block per (b, row-block rb): 16 rows x 128 cols.  Each thread owns 8
// consecutive pixels (one 8-px run inside a single 16-col block), so its
// private max feeds both the row-block max and the col-block partial max.
// ---------------------------------------------------------------------------
__global__ void k1_conv1_pool(const float* __restrict__ x,
                              const float* __restrict__ w1,
                              const float* __restrict__ g1, const float* __restrict__ b1,
                              const float* __restrict__ m1, const float* __restrict__ v1,
                              float* __restrict__ rp,      // [B,G,8]
                              float* __restrict__ cpart)   // [B,G,8(rb),8(cb)]
{
    const int blk = blockIdx.x;
    const int b  = blk >> 3;
    const int rb = blk & 7;
    const int t  = threadIdx.x;

    float acc[Gn][8];
#pragma unroll
    for (int g = 0; g < Gn; ++g)
#pragma unroll
        for (int j = 0; j < 8; ++j) acc[g][j] = 0.f;

    const long xbase = (long)b * Cn * HW + rb * 16 * Wd + t * 8;
    for (int c = 0; c < Cn; ++c) {
        const float4 xa = *reinterpret_cast<const float4*>(x + xbase + (long)c * HW);
        const float4 xb = *reinterpret_cast<const float4*>(x + xbase + (long)c * HW + 4);
        float xv[8] = {xa.x, xa.y, xa.z, xa.w, xb.x, xb.y, xb.z, xb.w};
#pragma unroll
        for (int g = 0; g < Gn; ++g) {
            const float wgt = w1[g * Cn + c];
#pragma unroll
            for (int j = 0; j < 8; ++j) acc[g][j] = fmaf(wgt, xv[j], acc[g][j]);
        }
    }

    // BN + ReLU, per-thread max
    float tmax[Gn];
#pragma unroll
    for (int g = 0; g < Gn; ++g) {
        const float inv  = g1[g] * rsqrtf(v1[g] + EPSV);
        const float beta = b1[g] - m1[g] * inv;
        float mg = -3.402823466e38f;
#pragma unroll
        for (int j = 0; j < 8; ++j) {
            float v = fmaxf(0.f, fmaf(acc[g][j], inv, beta));
            mg = fmaxf(mg, v);
        }
        tmax[g] = mg;
    }

    __shared__ float red[Gn * 256];
    __shared__ float cpm[Gn * 8];
#pragma unroll
    for (int g = 0; g < Gn; ++g) red[g * 256 + t] = tmax[g];
    __syncthreads();

    // col-block partials: threads with (t%16)>>1 == cb contribute
    if (t < 32) {
        const int g  = t >> 3;
        const int cb = t & 7;
        float m = -3.402823466e38f;
#pragma unroll
        for (int s = 0; s < 32; ++s) {
            const int tt = (cb * 2 + (s & 1)) + 16 * (s >> 1);
            m = fmaxf(m, red[g * 256 + tt]);
        }
        cpm[t] = m;
        cpart[((b * Gn + g) * 8 + rb) * 8 + cb] = m;
    }
    __syncthreads();

    // row-block max = max over the 8 col partials
    if (t < Gn) {
        float m = cpm[t * 8];
#pragma unroll
        for (int cb = 1; cb < 8; ++cb) m = fmaxf(m, cpm[t * 8 + cb]);
        rp[(b * Gn + t) * 8 + rb] = m;
    }
}

// ---------------------------------------------------------------------------
// Kernel prep: conv2_w -> bf16, BN2 scale/bias
// ---------------------------------------------------------------------------
__global__ void k_prep(const float* __restrict__ w2,
                       const float* __restrict__ g2, const float* __restrict__ b2,
                       const float* __restrict__ m2, const float* __restrict__ v2,
                       unsigned short* __restrict__ wbf,
                       float* __restrict__ alpha2, float* __restrict__ beta2)
{
    const int idx = blockIdx.x * 256 + threadIdx.x;
    if (idx < Cn * Cn) wbf[idx] = (unsigned short)f32_to_bf16_bits(w2[idx]);
    if (blockIdx.x == 0 && threadIdx.x < Cn) {
        const float a = g2[threadIdx.x] * rsqrtf(v2[threadIdx.x] + EPSV);
        alpha2[threadIdx.x] = a;
        beta2[threadIdx.x]  = b2[threadIdx.x] - m2[threadIdx.x] * a;
    }
}

// ---------------------------------------------------------------------------
// Kernel 2: p/q attention.  One block per batch, thread o in [0,256).
// ---------------------------------------------------------------------------
__global__ void k2_pq(const float* __restrict__ rp, const float* __restrict__ cpart,
                      const float* __restrict__ wp, const float* __restrict__ bp,
                      const float* __restrict__ wq, const float* __restrict__ bq,
                      float* __restrict__ pn, float* __restrict__ qn)
{
    const int b = blockIdx.x;
    const int o = threadIdx.x;
    __shared__ float sr[32], sc[32], sp[256], sq[256];

    if (o < 32) {
        sr[o] = rp[b * 32 + o];
        float m = -3.402823466e38f;
#pragma unroll
        for (int rb = 0; rb < 8; ++rb)
            m = fmaxf(m, cpart[b * 256 + (o >> 3) * 64 + rb * 8 + (o & 7)]);
        sc[o] = m;
    }
    __syncthreads();

    float ap = bp[o], aq = bq[o];
#pragma unroll
    for (int gk = 0; gk < 32; ++gk) {
        ap = fmaf(wp[o * 32 + gk], sr[gk], ap);
        aq = fmaf(wq[o * 32 + gk], sc[gk], aq);
    }
    sp[o] = 1.f / (1.f + expf(-ap));
    sq[o] = 1.f / (1.f + expf(-aq));
    __syncthreads();

    // p: row-normalize over j (last axis); q: col-normalize over i
    {
        const int rowbase = o & ~7;
        float s = 0.f;
#pragma unroll
        for (int j = 0; j < 8; ++j) s += sp[rowbase + j];
        pn[b * 256 + o] = sp[o] / s;
    }
    {
        const int g = o >> 6, j = o & 7;
        float s = 0.f;
#pragma unroll
        for (int i = 0; i < 8; ++i) s += sq[g * 64 + i * 8 + j];
        qn[b * 256 + o] = sq[o] / s;
    }
}

// ---------------------------------------------------------------------------
// Kernel 3: bilinear block mixing for one (b, ch).  Row-mix into a full
// 128x128 LDS image, then col-mix.  Each thread handles 2 adjacent pixels:
// float2 reads (256B/wave) and packed 2xbf16 u32 stores (128B/wave).
// ---------------------------------------------------------------------------
__global__ void k3_mix(const float* __restrict__ x,
                       const float* __restrict__ pn, const float* __restrict__ qn,
                       unsigned short* __restrict__ zb)
{
    const int b  = blockIdx.x >> 7;
    const int ch = blockIdx.x & 127;
    const int g  = ch / CG;
    const int t  = threadIdx.x;

    __shared__ float ps[64], qs[64];
    __shared__ float y[HW];          // 64 KB of the 320 KB WGP LDS

    if (t < 64)       ps[t]      = pn[b * 256 + g * 64 + t];
    else if (t < 128) qs[t - 64] = qn[b * 256 + g * 64 + (t - 64)];
    __syncthreads();

    const float* xim = x + ((long)b * Cn + ch) * HW;

    // row mix: y[i*16+a, w] = sum_jr p[i,jr] * x[jr*16+a, w]   (2 px/thread)
#pragma unroll 1
    for (int it = 0; it < 32; ++it) {
        const int pid = (t + 256 * it) * 2;
        const int h = pid >> 7, w = pid & 127;
        const int i = h >> 4,  a = h & 15;
        float ax = 0.f, ay = 0.f;
#pragma unroll
        for (int jr = 0; jr < 8; ++jr) {
            const float2 xv = *reinterpret_cast<const float2*>(
                xim + (jr * 16 + a) * Wd + w);
            const float pw = ps[i * 8 + jr];
            ax = fmaf(pw, xv.x, ax);
            ay = fmaf(pw, xv.y, ay);
        }
        *reinterpret_cast<float2*>(&y[pid]) = make_float2(ax, ay);
    }
    __syncthreads();

    // col mix: z[h, jc*16+v] = sum_ic q[ic,jc] * y[h, ic*16+v]  (2 px/thread)
    unsigned short* zim = zb + ((long)b * Cn + ch) * HW;
#pragma unroll 1
    for (int it = 0; it < 32; ++it) {
        const int pid = (t + 256 * it) * 2;
        const int h = pid >> 7, w = pid & 127;
        const int jc = w >> 4, v = w & 15;        // v is even
        float ax = 0.f, ay = 0.f;
#pragma unroll
        for (int ic = 0; ic < 8; ++ic) {
            const float qw = qs[ic * 8 + jc];
            const int   yi = h * Wd + ic * 16 + v;
            ax = fmaf(qw, y[yi],     ax);
            ay = fmaf(qw, y[yi + 1], ay);
        }
        const unsigned int packed =
            f32_to_bf16_bits(ax) | (f32_to_bf16_bits(ay) << 16);
        *reinterpret_cast<unsigned int*>(zim + pid) = packed;
    }
}

// ---------------------------------------------------------------------------
// Kernel 4: conv2 GEMM with bf16 WMMA + BN + ReLU.
// Block = 8 waves, one batch, 128 spatial columns.  LDS pitch 136 (+8 pad)
// gives a 272B row stride -> bank-rotated, conflict-free fragment reads.
// Fragment layouts per ISA 7.12.2 (16-bit A 16x32, B 32x16, f32 C/D).
// ---------------------------------------------------------------------------
#define LP 136
__global__ void k4_conv2_wmma(const unsigned short* __restrict__ zb,
                              const unsigned short* __restrict__ wbf,
                              const float* __restrict__ alpha2,
                              const float* __restrict__ beta2,
                              float* __restrict__ out)
{
    const int b  = blockIdx.x >> 7;
    const int n0 = (blockIdx.x & 127) * 128;
    const int t    = threadIdx.x;
    const int wave = t >> 5;
    const int lane = t & 31;
    const int lhalf = (lane >= 16);
    const int l16   = lane & 15;

    __shared__ unsigned short ldsW[Cn * LP];   // weights   [o][k]
    __shared__ unsigned short ldsZ[Cn * LP];   // z-tile^T  [n][c]
    __shared__ float sA[Cn], sB[Cn];

    if (t < Cn) { sA[t] = alpha2[t]; sB[t] = beta2[t]; }

    // stage weights: 16384 bf16 as 2048 uint4 vectors
#pragma unroll
    for (int it = 0; it < 8; ++it) {
        const int vec = t + 256 * it;
        const int e = vec * 8, o = e >> 7, k = e & 127;
        const uint4 d = *reinterpret_cast<const uint4*>(wbf + o * Cn + k);
        *reinterpret_cast<uint4*>(&ldsW[o * LP + k]) = d;
    }
    // stage z tile transposed: read 8 contiguous n, scatter into [n][c]
#pragma unroll
    for (int it = 0; it < 8; ++it) {
        const int vec = t + 256 * it;
        const int e = vec * 8, c = e >> 7, nn = e & 127;
        const uint4 d = *reinterpret_cast<const uint4*>(
            zb + ((long)b * Cn + c) * HW + n0 + nn);
        const unsigned short* u = reinterpret_cast<const unsigned short*>(&d);
#pragma unroll
        for (int j = 0; j < 8; ++j) ldsZ[(nn + j) * LP + c] = u[j];
    }
    __syncthreads();

    v8f acc[8];
#pragma unroll
    for (int m = 0; m < 8; ++m) acc[m] = v8f{};

    const int nloc = wave * 16 + l16;          // local column owned by this lane
#pragma unroll
    for (int ks = 0; ks < 4; ++ks) {
        const int k0 = ks * 32;
        FragAB fb;                             // B 32x16: lane holds 16 K of col n
        {
            const int idx = nloc * LP + k0 + (lhalf ? 8 : 0);
            fb.f[0] = *reinterpret_cast<const float4*>(&ldsZ[idx]);
            fb.f[1] = *reinterpret_cast<const float4*>(&ldsZ[idx + 16]);
        }
#pragma unroll
        for (int m = 0; m < 8; ++m) {
            FragAB fa;                         // A 16x32: lane holds 16 K of row o
            const int orow = m * 16 + l16;
            const int idx = orow * LP + k0 + (lhalf ? 8 : 0);
            fa.f[0] = *reinterpret_cast<const float4*>(&ldsW[idx]);
            fa.f[1] = *reinterpret_cast<const float4*>(&ldsW[idx + 16]);
            acc[m] = __builtin_amdgcn_wmma_f32_16x16x32_bf16(
                false, fa.v, false, fb.v, (short)0, acc[m], false, false);
        }
    }

    // epilogue: C/D layout — VGPR r holds (M = r + 8*lhalf, N = lane&15)
    const int n = n0 + nloc;
#pragma unroll
    for (int m = 0; m < 8; ++m) {
#pragma unroll
        for (int r = 0; r < 8; ++r) {
            const int o = m * 16 + r + (lhalf ? 8 : 0);
            const float v = fmaxf(0.f, fmaf(acc[m][r], sA[o], sB[o]));
            out[((long)b * Cn + o) * HW + n] = v;
        }
    }
}

// ---------------------------------------------------------------------------
extern "C" void kernel_launch(void* const* d_in, const int* in_sizes, int n_in,
                              void* d_out, int out_size, void* d_ws, size_t ws_size,
                              hipStream_t stream) {
    const float* x    = (const float*)d_in[0];
    const float* w1   = (const float*)d_in[1];
    const float* g1   = (const float*)d_in[2];
    const float* b1   = (const float*)d_in[3];
    const float* m1   = (const float*)d_in[4];
    const float* v1   = (const float*)d_in[5];
    const float* wp   = (const float*)d_in[6];
    const float* bp   = (const float*)d_in[7];
    const float* wq   = (const float*)d_in[8];
    const float* bq   = (const float*)d_in[9];
    const float* w2   = (const float*)d_in[10];
    const float* g2   = (const float*)d_in[11];
    const float* b2   = (const float*)d_in[12];
    const float* m2   = (const float*)d_in[13];
    const float* v2   = (const float*)d_in[14];
    float* out = (float*)d_out;

    // workspace carve-up (float elements)
    float* ws      = (float*)d_ws;
    float* rp      = ws;                 // 512
    float* cpart   = ws + 512;           // 4096
    float* pn      = ws + 4608;          // 4096
    float* qn      = ws + 8704;          // 4096
    float* alpha2  = ws + 12800;         // 128
    float* beta2   = ws + 12928;         // 128
    unsigned short* wbf = (unsigned short*)(ws + 13056);   // 16384 ushort
    unsigned short* zb  = (unsigned short*)(ws + 21248);   // B*C*HW ushort (64 MB)

    k1_conv1_pool<<<dim3(Bn * 8), dim3(256), 0, stream>>>(
        x, w1, g1, b1, m1, v1, rp, cpart);
    k_prep<<<dim3(64), dim3(256), 0, stream>>>(
        w2, g2, b2, m2, v2, wbf, alpha2, beta2);
    k2_pq<<<dim3(Bn), dim3(256), 0, stream>>>(
        rp, cpart, wp, bp, wq, bq, pn, qn);
    k3_mix<<<dim3(Bn * Cn), dim3(256), 0, stream>>>(
        x, pn, qn, zb);
    k4_conv2_wmma<<<dim3(Bn * 128), dim3(256), 0, stream>>>(
        zb, wbf, alpha2, beta2, out);
}
